// AttentivePoolingNetwork_54116587929808
// MI455X (gfx1250) — matrix operand order repaired and checked by reading, not compiled, not run on hardware
//
#include <hip/hip_runtime.h>
#include <stddef.h>

typedef unsigned short ushort_t;
typedef __attribute__((ext_vector_type(16))) __bf16 v16bf;
typedef __attribute__((ext_vector_type(8)))  float  v8f;

#define E_DIM 300
#define F_DIM 900
#define FPAD  928      // 29 * 32
#define C_DIM 400
#define CPAD  416      // 13 * 32
#define B_SZ  256
#define M_Q   40
#define L_A   400

// ---------- bf16 helpers (bit-level, round-to-nearest-even) ----------
__device__ __forceinline__ float bf2f(ushort_t u) {
    unsigned x = ((unsigned)u) << 16;
    float f; __builtin_memcpy(&f, &x, 4); return f;
}
__device__ __forceinline__ ushort_t f2bf(float f) {
    unsigned x; __builtin_memcpy(&x, &f, 4);
    unsigned r = x + 0x7FFFu + ((x >> 16) & 1u);
    return (ushort_t)(r >> 16);
}

// ---------- CDNA5 async global->LDS copy (ASYNCcnt-tracked, ISA 08 §4) ----------
// VDST = per-lane LDS byte address, VADDR = per-lane 64-bit global address.
__device__ __forceinline__ void async_load_b128(const void* gptr, void* lptr) {
    unsigned lds = (unsigned)(unsigned long long)lptr;   // LDS offset = low 32 bits
    asm volatile("global_load_async_to_lds_b128 %0, %1, off"
                 :: "v"(lds), "v"((unsigned long long)gptr)
                 : "memory");
}
__device__ __forceinline__ void wait_async0() {
    asm volatile("s_wait_asynccnt 0x0" ::: "memory");
}

// ---------- im2col: Z[(b,t), f] = x[b, t + f/E - 1, f%E]  (zero padded) ----------
__global__ void build_Z(const float* __restrict__ x, ushort_t* __restrict__ Z,
                        int T, long long total) {
    long long idx = (long long)blockIdx.x * blockDim.x + threadIdx.x;
    if (idx >= total) return;
    int f   = (int)(idx % FPAD);
    long long row = idx / FPAD;
    int t = (int)(row % T);
    int b = (int)(row / T);
    float v = 0.f;
    if (f < F_DIM) {
        int dt = f / E_DIM - 1;
        int e  = f - (dt + 1) * E_DIM;
        int tt = t + dt;
        if (tt >= 0 && tt < T) v = x[((size_t)b * T + tt) * E_DIM + e];
    }
    Z[idx] = f2bf(v);
}

// ---------- weight convert with K padding ----------
__global__ void build_W(const float* __restrict__ W, ushort_t* __restrict__ Wb) {
    int idx = blockIdx.x * blockDim.x + threadIdx.x;
    if (idx >= C_DIM * FPAD) return;
    int c = idx / FPAD, f = idx % FPAD;
    Wb[idx] = f2bf(f < F_DIM ? W[c * F_DIM + f] : 0.f);
}

// ---------- U transpose + pad: Ut[c, d] = d<400 ? U[d, c] : 0 ----------
__global__ void build_Ut(const float* __restrict__ U, ushort_t* __restrict__ Ut) {
    int idx = blockIdx.x * blockDim.x + threadIdx.x;
    if (idx >= C_DIM * CPAD) return;
    int c = idx / CPAD, d = idx % CPAD;
    Ut[idx] = f2bf(d < C_DIM ? U[d * C_DIM + c] : 0.f);
}

__global__ void zero_u16(ushort_t* __restrict__ p, long long n) {
    long long i = (long long)blockIdx.x * blockDim.x + threadIdx.x;
    if (i < n) p[i] = 0;
}

// =====================================================================
// Tiled bf16 NT GEMM: O[r,c] = sum_k A[r,k]*B[c,k] + bias[c]  (bf16 out)
// Block tile 128x128, 8 waves (4 row-groups x 2 col-groups), wave tile
// 32x64 = 8 WMMA tiles per 32-K chunk. Double-buffered LDS fed by
// global_load_async_to_lds_b128; one s_wait_asynccnt + barrier per chunk.
// Fragment layouts per CDNA5 ISA 7.12.2 (wave32).
// =====================================================================
__global__ __launch_bounds__(256)
void gemm_bf16_nt(const ushort_t* __restrict__ A, const ushort_t* __restrict__ Bm,
                  const float* __restrict__ bias, ushort_t* __restrict__ Obf,
                  int Mtot, int N, int K, int lda, int ldb, int ldo) {
    __shared__ __align__(16) ushort_t lA[2][128 * 32];
    __shared__ __align__(16) ushort_t lB[2][128 * 32];

    const int t    = threadIdx.x;
    const int lane = t & 31;
    const int wv   = t >> 5;
    const int wm   = wv >> 1;        // 0..3 : 32-row group
    const int wn   = wv & 1;         // 0..1 : 64-col group
    const int half = lane >> 4;
    const int l16  = lane & 15;
    const int tileM = blockIdx.y * 128;
    const int tileN = blockIdx.x * 128;

    // staging coordinates: 128x32 tile = 4096 elems = 2 x b128 per thread
    const int r0  = (t * 8) >> 5,         kk0 = (t * 8) & 31;
    const int r1  = ((t + 256) * 8) >> 5, kk1 = ((t + 256) * 8) & 31;
    int ga0 = tileM + r0; if (ga0 > Mtot - 1) ga0 = Mtot - 1;
    int ga1 = tileM + r1; if (ga1 > Mtot - 1) ga1 = Mtot - 1;
    int gb0 = tileN + r0; if (gb0 > N - 1)    gb0 = N - 1;
    int gb1 = tileN + r1; if (gb1 > N - 1)    gb1 = N - 1;

    auto stage = [&](int k0, int buf) {
        async_load_b128(A  + (size_t)ga0 * lda + k0 + kk0, &lA[buf][r0 * 32 + kk0]);
        async_load_b128(A  + (size_t)ga1 * lda + k0 + kk1, &lA[buf][r1 * 32 + kk1]);
        async_load_b128(Bm + (size_t)gb0 * ldb + k0 + kk0, &lB[buf][r0 * 32 + kk0]);
        async_load_b128(Bm + (size_t)gb1 * ldb + k0 + kk1, &lB[buf][r1 * 32 + kk1]);
    };

    v8f acc[2][4];
#pragma unroll
    for (int i = 0; i < 2; ++i)
#pragma unroll
        for (int j = 0; j < 4; ++j)
#pragma unroll
            for (int v = 0; v < 8; ++v) acc[i][j][v] = 0.f;

    stage(0, 0);
    wait_async0();
    __syncthreads();

    int cur = 0;
    for (int k0 = 0; k0 < K; k0 += 32) {
        if (k0 + 32 < K) stage(k0 + 32, cur ^ 1);   // overlap next chunk

        union fragu { uint4 q[2]; v16bf v; };
        fragu fa[2], fb[4];
#pragma unroll
        for (int rt = 0; rt < 2; ++rt) {
            const uint4* pa = reinterpret_cast<const uint4*>(
                &lA[cur][(wm * 32 + rt * 16 + l16) * 32]);
            fa[rt].q[0] = pa[half];
            fa[rt].q[1] = pa[2 + half];
        }
#pragma unroll
        for (int nt = 0; nt < 4; ++nt) {
            const uint4* pb = reinterpret_cast<const uint4*>(
                &lB[cur][(wn * 64 + nt * 16 + l16) * 32]);
            fb[nt].q[0] = pb[half];
            fb[nt].q[1] = pb[2 + half];
        }
#pragma unroll
        for (int rt = 0; rt < 2; ++rt)
#pragma unroll
            for (int nt = 0; nt < 4; ++nt)
                acc[rt][nt] = __builtin_amdgcn_wmma_f32_16x16x32_bf16(
                    false, fa[rt].v, false, fb[nt].v, (short)0, acc[rt][nt], false, false);

        wait_async0();       // next buffer resident for this wave
        __syncthreads();     // ... and for all waves; cur safe to overwrite
        cur ^= 1;
    }

#pragma unroll
    for (int rt = 0; rt < 2; ++rt)
#pragma unroll
        for (int nt = 0; nt < 4; ++nt) {
            int gc = tileN + wn * 64 + nt * 16 + l16;
            if (gc >= N) continue;
            float bv = bias ? bias[gc] : 0.f;
#pragma unroll
            for (int v = 0; v < 8; ++v) {
                int gr = tileM + wm * 32 + rt * 16 + v + 8 * half;
                if (gr < Mtot) Obf[(size_t)gr * ldo + gc] = f2bf(acc[rt][nt][v] + bv);
            }
        }
}

// =====================================================================
// Batched: G[b, m, l] = tanh( sum_c T[b*40+m, c] * Abf[b*400+l, c] )
// Block tile 48 rows x 128 cols per batch; wave w owns col tile w (16),
// 3 row tiles each. K = CPAD = 416. Async-LDS staging.
// =====================================================================
__global__ __launch_bounds__(256)
void gemm_batched_tanh(const ushort_t* __restrict__ Tbf, const ushort_t* __restrict__ Abf,
                       float* __restrict__ G) {
    __shared__ __align__(16) ushort_t lA[48 * 32];
    __shared__ __align__(16) ushort_t lB[128 * 32];

    const int t    = threadIdx.x;
    const int lane = t & 31;
    const int wv   = t >> 5;         // 0..7 -> col tile
    const int half = lane >> 4;
    const int l16  = lane & 15;
    const int b    = blockIdx.z;
    const int tileL = blockIdx.x * 128;

    const int ra  = (t * 8) >> 5,         ka  = (t * 8) & 31;          // A: t<192
    const int rb0 = (t * 8) >> 5,         kb0 = (t * 8) & 31;          // B part 0
    const int rb1 = ((t + 256) * 8) >> 5, kb1 = ((t + 256) * 8) & 31;  // B part 1
    int gl0 = tileL + rb0; if (gl0 > L_A - 1) gl0 = L_A - 1;
    int gl1 = tileL + rb1; if (gl1 > L_A - 1) gl1 = L_A - 1;

    v8f acc[3];
#pragma unroll
    for (int i = 0; i < 3; ++i)
#pragma unroll
        for (int j = 0; j < 8; ++j) acc[i][j] = 0.f;

    for (int k0 = 0; k0 < CPAD; k0 += 32) {
        __syncthreads();             // prior reads done before overwrite
        if (t < 192)                 // stage A 48x32 (Tbf has >=16 slack rows)
            async_load_b128(Tbf + (size_t)(b * M_Q + ra) * CPAD + k0 + ka,
                            &lA[ra * 32 + ka]);
        async_load_b128(Abf + ((size_t)b * L_A + gl0) * CPAD + k0 + kb0,
                        &lB[rb0 * 32 + kb0]);
        async_load_b128(Abf + ((size_t)b * L_A + gl1) * CPAD + k0 + kb1,
                        &lB[rb1 * 32 + kb1]);
        wait_async0();
        __syncthreads();

        union fragu { uint4 q[2]; v16bf v; };
        fragu ub, fa[3];
        const uint4* pb = reinterpret_cast<const uint4*>(&lB[(wv * 16 + l16) * 32]);
        ub.q[0] = pb[half];
        ub.q[1] = pb[2 + half];
#pragma unroll
        for (int rt = 0; rt < 3; ++rt) {
            const uint4* pa = reinterpret_cast<const uint4*>(&lA[(rt * 16 + l16) * 32]);
            fa[rt].q[0] = pa[half];
            fa[rt].q[1] = pa[2 + half];
        }
#pragma unroll
        for (int rt = 0; rt < 3; ++rt)
            acc[rt] = __builtin_amdgcn_wmma_f32_16x16x32_bf16(
                false, fa[rt].v, false, ub.v, (short)0, acc[rt], false, false);
    }

    int l = tileL + wv * 16 + l16;
    if (l >= L_A) return;
#pragma unroll
    for (int rt = 0; rt < 3; ++rt) {
#pragma unroll
        for (int v = 0; v < 8; ++v) {
            int m = rt * 16 + v + 8 * half;
            if (m < M_Q)
                G[((size_t)b * M_Q + m) * L_A + l] = tanhf(acc[rt][v]);
        }
    }
}

// ---------- reductions ----------
__global__ void rowmax_G(const float* __restrict__ G, float* __restrict__ gq) {
    int id = blockIdx.x * blockDim.x + threadIdx.x;      // B*M rows
    if (id >= B_SZ * M_Q) return;
    size_t base = (size_t)id * L_A;
    float m = -2.f;
    for (int l = 0; l < L_A; ++l) m = fmaxf(m, G[base + l]);
    gq[id] = m;
}

__global__ void colmax_G(const float* __restrict__ G, float* __restrict__ ga) {
    int id = blockIdx.x * blockDim.x + threadIdx.x;      // B*L entries
    if (id >= B_SZ * L_A) return;
    int b = id / L_A, l = id % L_A;
    float m = -2.f;
    for (int mm = 0; mm < M_Q; ++mm)
        m = fmaxf(m, G[((size_t)b * M_Q + mm) * L_A + l]);
    ga[id] = m;
}

// softmax over BATCH dim: ro[b, col] = exp(g[b,col]) / sum_b' exp(g[b',col])
__global__ void softmax_batch(const float* __restrict__ g, float* __restrict__ ro, int NC) {
    int col = blockIdx.x;
    int t   = threadIdx.x;           // t == batch index (blockDim == 256 == B)
    __shared__ float s[B_SZ];
    float v = g[(size_t)t * NC + col];
    s[t] = v; __syncthreads();
    for (int off = 128; off > 0; off >>= 1) {
        if (t < off) s[t] = fmaxf(s[t], s[t + off]);
        __syncthreads();
    }
    float mx = s[0]; __syncthreads();
    float e = expf(v - mx);
    s[t] = e; __syncthreads();
    for (int off = 128; off > 0; off >>= 1) {
        if (t < off) s[t] += s[t + off];
        __syncthreads();
    }
    ro[(size_t)t * NC + col] = e / s[0];
}

// rQ[b,c] = sum_m Qbf[(b*40+m), c] * roQ[b*40+m]
__global__ void pool_rQ(const ushort_t* __restrict__ Qbf, const float* __restrict__ roQ,
                        float* __restrict__ rQ) {
    int id = blockIdx.x * blockDim.x + threadIdx.x;
    if (id >= B_SZ * C_DIM) return;
    int b = id / C_DIM, c = id % C_DIM;
    float s = 0.f;
    for (int m = 0; m < M_Q; ++m)
        s += bf2f(Qbf[((size_t)b * M_Q + m) * CPAD + c]) * roQ[b * M_Q + m];
    rQ[id] = s;
}

// rA[b,c] = sum_l Abf[(b*400+l), c] * roA[b*400+l]
__global__ void pool_rA(const ushort_t* __restrict__ Abf, const float* __restrict__ roA,
                        float* __restrict__ rA) {
    int id = blockIdx.x * blockDim.x + threadIdx.x;
    if (id >= B_SZ * C_DIM) return;
    int b = id / C_DIM, c = id % C_DIM;
    float s = 0.f;
    for (int l = 0; l < L_A; ++l)
        s += bf2f(Abf[((size_t)b * L_A + l) * CPAD + c]) * roA[b * L_A + l];
    rA[id] = s;
}

__global__ void cosine_out(const float* __restrict__ rQ, const float* __restrict__ rA,
                           float* __restrict__ out) {
    int b = blockIdx.x, t = threadIdx.x;
    __shared__ float sd[256], sq[256], sa[256];
    float d = 0.f, q = 0.f, a = 0.f;
    for (int c = t; c < C_DIM; c += 256) {
        float x = rQ[(size_t)b * C_DIM + c], y = rA[(size_t)b * C_DIM + c];
        d += x * y; q += x * x; a += y * y;
    }
    sd[t] = d; sq[t] = q; sa[t] = a; __syncthreads();
    for (int off = 128; off > 0; off >>= 1) {
        if (t < off) { sd[t] += sd[t + off]; sq[t] += sq[t + off]; sa[t] += sa[t + off]; }
        __syncthreads();
    }
    if (t == 0)
        out[b] = sd[0] / (fmaxf(sqrtf(sq[0]), 1e-6f) * fmaxf(sqrtf(sa[0]), 1e-6f));
}

// =====================================================================
extern "C" void kernel_launch(void* const* d_in, const int* in_sizes, int n_in,
                              void* d_out, int out_size, void* d_ws, size_t ws_size,
                              hipStream_t stream) {
    (void)in_sizes; (void)n_in; (void)out_size; (void)ws_size;
    const float* question = (const float*)d_in[0];   // [B, M, E]
    const float* answer   = (const float*)d_in[1];   // [B, L, E]
    const float* Wq       = (const float*)d_in[2];   // [C, F]
    const float* bq       = (const float*)d_in[3];   // [C]
    const float* Wa       = (const float*)d_in[4];   // [C, F]
    const float* ba       = (const float*)d_in[5];   // [C]
    const float* U        = (const float*)d_in[6];   // [C, C]
    float* out = (float*)d_out;                      // [B]

    // workspace carve-out
    size_t off = 0;
    auto alloc = [&](size_t bytes) -> void* {
        off = (off + 255) & ~(size_t)255;
        void* p = (char*)d_ws + off;
        off += bytes;
        return p;
    };
    const int BM = B_SZ * M_Q;               // 10240
    const int BL = B_SZ * L_A;               // 102400
    const int T_SLACK = 16;
    ushort_t* Zq  = (ushort_t*)alloc((size_t)BM * FPAD * 2);
    ushort_t* Za  = (ushort_t*)alloc((size_t)BL * FPAD * 2);
    ushort_t* Wqb = (ushort_t*)alloc((size_t)C_DIM * FPAD * 2);
    ushort_t* Wab = (ushort_t*)alloc((size_t)C_DIM * FPAD * 2);
    ushort_t* Utb = (ushort_t*)alloc((size_t)C_DIM * CPAD * 2);
    ushort_t* Qbf = (ushort_t*)alloc((size_t)BM * CPAD * 2);
    ushort_t* Abf = (ushort_t*)alloc((size_t)BL * CPAD * 2);
    ushort_t* Tbf = (ushort_t*)alloc((size_t)(BM + T_SLACK) * CPAD * 2);
    float*    G   = (float*)alloc((size_t)BM * L_A * 4);
    float*    gq  = (float*)alloc((size_t)BM * 4);
    float*    ga  = (float*)alloc((size_t)BL * 4);
    float*    roQ = (float*)alloc((size_t)BM * 4);
    float*    roA = (float*)alloc((size_t)BL * 4);
    float*    rQ  = (float*)alloc((size_t)B_SZ * C_DIM * 4);
    float*    rA  = (float*)alloc((size_t)B_SZ * C_DIM * 4);

    // 1) im2col + weight conversion
    {
        long long tq = (long long)BM * FPAD;
        long long ta = (long long)BL * FPAD;
        build_Z<<<(unsigned)((tq + 255) / 256), 256, 0, stream>>>(question, Zq, M_Q, tq);
        build_Z<<<(unsigned)((ta + 255) / 256), 256, 0, stream>>>(answer,   Za, L_A, ta);
        int nw = C_DIM * FPAD;
        build_W<<<(nw + 255) / 256, 256, 0, stream>>>(Wq, Wqb);
        build_W<<<(nw + 255) / 256, 256, 0, stream>>>(Wa, Wab);
        int nu = C_DIM * CPAD;
        build_Ut<<<(nu + 255) / 256, 256, 0, stream>>>(U, Utb);
        long long tz = (long long)(BM + T_SLACK) * CPAD;
        zero_u16<<<(unsigned)((tz + 255) / 256), 256, 0, stream>>>(Tbf, tz);
    }

    // 2) encodes: Q = Zq @ Wq^T + bq ; A = Za @ Wa^T + ba   (bf16, K-padded out)
    {
        dim3 g1((C_DIM + 127) / 128, BM / 128, 1);
        gemm_bf16_nt<<<g1, 256, 0, stream>>>(Zq, Wqb, bq, Qbf,
                                             BM, C_DIM, FPAD, FPAD, FPAD, CPAD);
        dim3 g2((C_DIM + 127) / 128, BL / 128, 1);
        gemm_bf16_nt<<<g2, 256, 0, stream>>>(Za, Wab, ba, Abf,
                                             BL, C_DIM, FPAD, FPAD, FPAD, CPAD);
    }

    // 3) T = Q @ U  (via U^T, NT form)
    {
        dim3 g3((C_DIM + 127) / 128, BM / 128, 1);
        gemm_bf16_nt<<<g3, 256, 0, stream>>>(Qbf, Utb, nullptr, Tbf,
                                             BM, C_DIM, CPAD, CPAD, CPAD, CPAD);
    }

    // 4) G = tanh(T_b @ A_b^T), batched
    {
        dim3 g4((L_A + 127) / 128, 1, B_SZ);
        gemm_batched_tanh<<<g4, 256, 0, stream>>>(Tbf, Abf, G);
    }

    // 5) pools + batch softmax + attention pooling + cosine
    rowmax_G<<<BM / 256, 256, 0, stream>>>(G, gq);
    colmax_G<<<BL / 256, 256, 0, stream>>>(G, ga);
    softmax_batch<<<M_Q, 256, 0, stream>>>(gq, roQ, M_Q);
    softmax_batch<<<L_A, 256, 0, stream>>>(ga, roA, L_A);
    pool_rQ<<<(B_SZ * C_DIM) / 256, 256, 0, stream>>>(Qbf, roQ, rQ);
    pool_rA<<<(B_SZ * C_DIM) / 256, 256, 0, stream>>>(Abf, roA, rA);
    cosine_out<<<B_SZ, 256, 0, stream>>>(rQ, rA, out);
}
